// LCSCC_29085518529166
// MI455X (gfx1250) — compile-verified, matrix operand order checked by reading, not run on hardware
//
#include <hip/hip_runtime.h>

// MI455X / gfx1250, wave32. GEMM via v_wmma_f32_16x16x32_bf16; B panels staged
// in LDS (shared by 4 waves of one batch), double-buffered with async copies.
typedef __attribute__((ext_vector_type(16))) __bf16 v16bf;
typedef __attribute__((ext_vector_type(8)))  __bf16 v8bf;
typedef __attribute__((ext_vector_type(8)))  float  v8f;
typedef int v4i __attribute__((vector_size(4 * sizeof(int))));

#define CCH    256
#define HDIM   64
#define WDIM   64
#define PPATCH 1024
#define KTILES 8            // 256 / 32
#define QTILES 64           // 1024 / 16
#define INV_2S2 (1.0f/5.12f)   // zero-window: sr=sc=1.6, 2*sr^2=5.12
#define ROWP   264          // padded LDS row stride in halves (528 B): bank-conflict free

#if defined(__AMDGCN__) && __has_builtin(__builtin_amdgcn_global_load_async_to_lds_b128)
#define USE_ASYNC 1
#else
#define USE_ASYNC 0
#endif

// AS1 pointer to a 16B vector (clang prints AS1 as "__device__" in diagnostics)
#define AS1P(p) ((__attribute__((address_space(1))) v4i*)(uintptr_t)(p))
// flat LDS address[31:0] == LDS byte offset (aperture rule), so truncate+inttoptr
#define AS3P(p) ((__attribute__((address_space(3))) v4i*)(uintptr_t)(unsigned)(uintptr_t)(p))

__device__ __forceinline__ unsigned short f2bf(float f) {
  unsigned u = __float_as_uint(f);
  u += 0x7FFFu + ((u >> 16) & 1u);      // round-to-nearest-even
  return (unsigned short)(u >> 16);
}

// ---------------- kernel 0: L2-normalize over channels, pack G[ba][p][c] bf16
__global__ void k_pack(const float* __restrict__ x, unsigned short* __restrict__ G) {
  int tid = blockIdx.x * blockDim.x + threadIdx.x;     // 0 .. 32767 pixels
  int b   = tid >> 12;
  int rem = tid & 4095;
  int y   = rem >> 6, xc = rem & 63;
  const float* px = x + (size_t)b * CCH * 4096 + y * WDIM + xc;
  float ss = 0.f;
  #pragma unroll 4
  for (int c = 0; c < CCH; ++c) { float v = px[(size_t)c * 4096]; ss += v * v; }
  float inv = 1.0f / fmaxf(sqrtf(ss), 1e-12f);
  int a = ((y & 1) << 1) | (xc & 1);
  int p = ((y >> 1) << 5) | (xc >> 1);
  unsigned short* dst = G + ((size_t)(b * 4 + a) * PPATCH + p) * CCH;
  #pragma unroll 4
  for (int c = 0; c < CCH; ++c) dst[c] = f2bf(px[(size_t)c * 4096] * inv);
}

// Stage one 16-row B panel (16 x 512B global, contiguous) into padded LDS rows.
// 512 16B chunks over 128 threads -> 4 async b128 copies per thread.
__device__ __forceinline__ void stage_tile(const unsigned short* __restrict__ gsrc,
                                           unsigned short* lbuf, int tid) {
  #pragma unroll
  for (int i = 0; i < 4; ++i) {
    int c   = tid + i * 128;
    int row = c >> 5, col = c & 31;
    int go  = row * 512 + col * 16;   // bytes in global panel
    int lo  = row * 528 + col * 16;   // bytes in padded LDS panel
#if USE_ASYNC
    __builtin_amdgcn_global_load_async_to_lds_b128(
        AS1P((const char*)gsrc + go), AS3P((char*)lbuf + lo), 0, 0);
#else
    *(uint4*)((char*)lbuf + lo) = *(const uint4*)((const char*)gsrc + go);
#endif
  }
}

__device__ __forceinline__ void wait_async0() {
#if USE_ASYNC
  asm volatile("s_wait_asynccnt 0x0" ::: "memory");
#endif
}

// A-fragment load from global (row-major G). bf16 A/B layouts are identical for
// G*G^T: lane L -> row base+(L&15); K-chunks {k..k+7,k+16..k+23} (lanes<16) or
// {k+8..k+15,k+24..k+31} (lanes>=16). Two contiguous 16B loads.
__device__ __forceinline__ v16bf load_frag(const __bf16* __restrict__ row,
                                           int kbase, int hi8) {
  union { v16bf v; v8bf h[2]; } u;
  u.h[0] = *(const v8bf*)(row + kbase + hi8);
  u.h[1] = *(const v8bf*)(row + kbase + 16 + hi8);
  return u.v;
}

__device__ __forceinline__ v16bf lds_frag(const unsigned short* __restrict__ lrow,
                                          int kbase, int hi8) {
  union { v16bf v; v8bf h[2]; } u;
  u.h[0] = *(const v8bf*)(lrow + kbase + hi8);        // ds_load_b128
  u.h[1] = *(const v8bf*)(lrow + kbase + 16 + hi8);
  return u.v;
}

// softmax is shift-invariant and |sa| <= alpha (=5): exp can't overflow fp32,
// so row stats reduce to plain sums R_p = sum_q exp(sa[p][q]).

// ---------------- kernel A: row softmax denominators
__global__ void __launch_bounds__(128)
k_stats(const __bf16* __restrict__ G, const float* __restrict__ alphap,
        float* __restrict__ Sr) {
  __shared__ alignas(16) unsigned short sbuf[2][16 * ROWP];
  __shared__ float gtab[32];
  int tid = threadIdx.x;
  if (tid < 32) { float d = (float)tid; gtab[tid] = __expf(-d * d * INV_2S2); }

  const float alpha = alphap[0];
  int lane = tid & 31;
  int wav  = tid >> 5;
  int ba   = blockIdx.x >> 4;                       // 32 batches x 16 blocks
  int p0   = (((blockIdx.x & 15) << 2) + wav) << 4; // 4 waves = 4 row slabs
  int l16  = lane & 15;
  int hi8  = (lane >> 4) << 3;
  const __bf16* Gb = G + (size_t)ba * PPATCH * CCH;

  v16bf af[KTILES];
  {
    const __bf16* Arow = Gb + (size_t)(p0 + l16) * CCH;
    #pragma unroll
    for (int f = 0; f < KTILES; ++f) af[f] = load_frag(Arow, f * 32, hi8);
  }

  int prow_base = p0 + hi8;
  int py[8], pxl[8];
  float R[8];
  #pragma unroll
  for (int r = 0; r < 8; ++r) {
    py[r] = (prow_base + r) >> 5; pxl[r] = (prow_base + r) & 31; R[r] = 0.f;
  }

  stage_tile((const unsigned short*)Gb, sbuf[0], tid);
  wait_async0();
  __syncthreads();

  for (int qt = 0; qt < QTILES; ++qt) {
    int cur = qt & 1;
    if (qt + 1 < QTILES)
      stage_tile((const unsigned short*)(Gb + (size_t)(qt + 1) * 16 * CCH),
                 sbuf[cur ^ 1], tid);

    const unsigned short* lrow = &sbuf[cur][l16 * ROWP];
    v8f acc = {};
    #pragma unroll
    for (int f = 0; f < KTILES; ++f) {
      v16bf bf = lds_frag(lrow, f * 32, hi8);
      acc = __builtin_amdgcn_wmma_f32_16x16x32_bf16(
          false, af[f], false, bf, (short)0, acc, false, false);
    }

    int q  = qt * 16 + l16;
    int qy = q >> 5, qx = q & 31;
    #pragma unroll
    for (int r = 0; r < 8; ++r) {
      float mask = 1.0f - gtab[abs(qy - py[r])] * gtab[abs(qx - pxl[r])];
      R[r] += __expf(acc[r] * mask * alpha);
    }
    wait_async0();       // next LDS buffer fully written (my ASYNCcnt==0)
    __syncthreads();     // everyone done reading cur before it's overwritten
  }

  #pragma unroll
  for (int r = 0; r < 8; ++r) {
    #pragma unroll
    for (int off = 1; off < 16; off <<= 1) R[r] += __shfl_xor(R[r], off, 32);
  }
  if (l16 == 0) {
    #pragma unroll
    for (int r = 0; r < 8; ++r)
      Sr[(size_t)ba * PPATCH + prow_base + r] = R[r];
  }
}

__device__ __forceinline__ void ins3(float v, float& t0, float& t1, float& t2) {
  if (v > t2) {
    if (v > t1) {
      t2 = t1;
      if (v > t0) { t1 = t0; t0 = v; } else { t1 = v; }
    } else t2 = v;
  }
}

// ---------------- kernel B: s = softmax_row * softmax_col, per-row top-3
__global__ void __launch_bounds__(128)
k_topk(const __bf16* __restrict__ G, const float* __restrict__ alphap,
       const float* __restrict__ Sr, float* __restrict__ out) {
  __shared__ alignas(16) unsigned short sbuf[2][16 * ROWP];
  __shared__ float gtab[32];
  int tid = threadIdx.x;
  if (tid < 32) { float d = (float)tid; gtab[tid] = __expf(-d * d * INV_2S2); }

  const float alpha = alphap[0];
  int lane = tid & 31;
  int wav  = tid >> 5;
  int ba   = blockIdx.x >> 4;
  int p0   = (((blockIdx.x & 15) << 2) + wav) << 4;
  int l16  = lane & 15;
  int hi8  = (lane >> 4) << 3;
  const __bf16* Gb = G + (size_t)ba * PPATCH * CCH;

  v16bf af[KTILES];
  {
    const __bf16* Arow = Gb + (size_t)(p0 + l16) * CCH;
    #pragma unroll
    for (int f = 0; f < KTILES; ++f) af[f] = load_frag(Arow, f * 32, hi8);
  }

  int prow_base = p0 + hi8;
  int py[8], pxl[8];
  float iRp[8], t0[8], t1[8], t2[8];
  #pragma unroll
  for (int r = 0; r < 8; ++r) {
    py[r]  = (prow_base + r) >> 5;
    pxl[r] = (prow_base + r) & 31;
    iRp[r] = 1.0f / Sr[(size_t)ba * PPATCH + prow_base + r];
    t0[r] = t1[r] = t2[r] = -1.0f;
  }

  stage_tile((const unsigned short*)Gb, sbuf[0], tid);
  wait_async0();
  __syncthreads();

  for (int qt = 0; qt < QTILES; ++qt) {
    int cur = qt & 1;
    if (qt + 1 < QTILES)
      stage_tile((const unsigned short*)(Gb + (size_t)(qt + 1) * 16 * CCH),
                 sbuf[cur ^ 1], tid);

    const unsigned short* lrow = &sbuf[cur][l16 * ROWP];
    v8f acc = {};
    #pragma unroll
    for (int f = 0; f < KTILES; ++f) {
      v16bf bf = lds_frag(lrow, f * 32, hi8);
      acc = __builtin_amdgcn_wmma_f32_16x16x32_bf16(
          false, af[f], false, bf, (short)0, acc, false, false);
    }

    int q  = qt * 16 + l16;
    int qy = q >> 5, qx = q & 31;
    float iRq = 1.0f / Sr[(size_t)ba * PPATCH + q];
    #pragma unroll
    for (int r = 0; r < 8; ++r) {
      float mask = 1.0f - gtab[abs(qy - py[r])] * gtab[abs(qx - pxl[r])];
      float sa = acc[r] * mask * alpha;
      float s  = __expf(sa + sa) * iRp[r] * iRq;   // exp(sa)/Rp * exp(sa)/Rq
      ins3(s, t0[r], t1[r], t2[r]);
    }
    wait_async0();
    __syncthreads();
  }

  // butterfly-merge top-3 across the 16 lanes of each half
  #pragma unroll
  for (int r = 0; r < 8; ++r) {
    #pragma unroll
    for (int off = 1; off < 16; off <<= 1) {
      float o0 = __shfl_xor(t0[r], off, 32);   // snapshot partner BEFORE merging
      float o1 = __shfl_xor(t1[r], off, 32);
      float o2 = __shfl_xor(t2[r], off, 32);
      ins3(o0, t0[r], t1[r], t2[r]);
      ins3(o1, t0[r], t1[r], t2[r]);
      ins3(o2, t0[r], t1[r], t2[r]);
    }
  }
  if (l16 == 0) {
    int b = ba >> 2, a = ba & 3;
    int ay = a >> 1, ax = a & 1;
    #pragma unroll
    for (int r = 0; r < 8; ++r) {
      int prow = prow_base + r;
      int y  = ((prow >> 5) << 1) | ay;
      int xx = ((prow & 31) << 1) | ax;
      size_t o = (((size_t)b * 3) * HDIM + y) * WDIM + xx;
      out[o]            = t0[r];
      out[o + 4096]     = t1[r];      // k=1 plane (64*64)
      out[o + 2 * 4096] = t2[r];      // k=2 plane
    }
  }
}

extern "C" void kernel_launch(void* const* d_in, const int* in_sizes, int n_in,
                              void* d_out, int out_size, void* d_ws, size_t ws_size,
                              hipStream_t stream) {
  const float* x     = (const float*)d_in[0];
  const float* alpha = (const float*)d_in[1];
  float* out = (float*)d_out;

  // workspace: G bf16 [32][1024][256] = 16 MiB, then row sums 128 KiB
  unsigned short* G = (unsigned short*)d_ws;
  float* Sr = (float*)((char*)d_ws + 16777216);

  k_pack <<<128, 256, 0, stream>>>(x, G);
  // 512 blocks = 32 batches x 16 blocks; 4 waves/block share one B panel in LDS
  k_stats<<<512, 128, 0, stream>>>((const __bf16*)G, alpha, Sr);
  k_topk <<<512, 128, 0, stream>>>((const __bf16*)G, alpha, Sr, out);
}
